// ClassBasedDecoder_28381143892585
// MI455X (gfx1250) — compile-verified, matrix operand order checked by reading, not run on hardware
//
#include <hip/hip_runtime.h>
#include <hip/hip_bf16.h>

// Problem constants (match reference)
#define NHID   1024
#define NWORDS 65536
#define NCLS   64
#define CSIZE  (NWORDS / NCLS)   // 1024 words per class
#define NTOK   8192
#define TPC    (NTOK / NCLS)     // 128 tokens per class

// Tiling
#define KC  32   // K-chunk staged in LDS == one WMMA K-step
#define LDA 40   // padded LDS row stride in bf16 (80B: 16B-aligned, bank-spread)

typedef __attribute__((ext_vector_type(16))) __bf16 v16bf;
typedef __attribute__((ext_vector_type(8)))  __bf16 v8bf;
typedef __attribute__((ext_vector_type(4)))  __bf16 v4bf;
typedef __attribute__((ext_vector_type(8)))  float  v8f;

// Load one 16x32 bf16 A/B fragment from an LDS plane per CDNA5 WMMA layout:
// lanes 0-15 hold K=0..7 (elems 0-7) and K=16..23 (elems 8-15);
// lanes 16-31 hold K=8..15 and K=24..31.
__device__ __forceinline__ v16bf frag_ld(const __bf16* plane, int row, int lane) {
  const __bf16* p = plane + row * LDA + ((lane >> 4) << 3);  // +0 or +8
  v8bf a = *(const v8bf*)(p);        // K base .. base+7   (16B aligned)
  v8bf b = *(const v8bf*)(p + 16);   // K base+16 .. +23
  return __builtin_shufflevector(a, b, 0,1,2,3,4,5,6,7,8,9,10,11,12,13,14,15);
}

// fp32 -> (hi, lo) bf16 split, stored to the two LDS planes (4 elems, 8B aligned)
__device__ __forceinline__ void cvt_store(__bf16* hi, __bf16* lo, int off, float4 v) {
  __bf16 h0 = (__bf16)v.x, h1 = (__bf16)v.y, h2 = (__bf16)v.z, h3 = (__bf16)v.w;
  v4bf h = {h0, h1, h2, h3};
  v4bf l = {(__bf16)(v.x - (float)h0), (__bf16)(v.y - (float)h1),
            (__bf16)(v.z - (float)h2), (__bf16)(v.w - (float)h3)};
  *(v4bf*)(hi + off) = h;
  *(v4bf*)(lo + off) = l;
}

__device__ __forceinline__ v8f wmma_bf16(v16bf a, v16bf b, v8f c) {
  return __builtin_amdgcn_wmma_f32_16x16x32_bf16(
      /*neg_a=*/false, a, /*neg_b=*/false, b,
      /*c_mod=*/(short)0, c, /*reuse_a=*/false, /*reuse_b=*/false);
}

// ---------------------------------------------------------------------------
// Grouped GEMM: p_words[c] = input[wbi[c]] @ word_emb[cluster[c]].T + bias
// grid = (CSIZE/128, NCLS), block = 256 (8 waves). Block tile: 128(M) x 128(N).
// ---------------------------------------------------------------------------
__global__ __launch_bounds__(256) void words_gemm_kernel(
    const float* __restrict__ input, const float* __restrict__ word_emb,
    const float* __restrict__ word_bias, const int* __restrict__ wbi,
    const int* __restrict__ cluster, float* __restrict__ out_words)
{
  __shared__ __bf16 Ahi[TPC * LDA], Alo[TPC * LDA];   // 10 KB each
  __shared__ __bf16 Bhi[128 * LDA], Blo[128 * LDA];   // 10 KB each  (40 KB total)

  const int c    = blockIdx.y;
  const int n0   = blockIdx.x * 128;
  const int tid  = threadIdx.x;
  const int lane = tid & 31;
  const int wave = tid >> 5;
  const int wm   = wave & 3;   // 4 M-strips of 32
  const int wn   = wave >> 2;  // 2 N-strips of 64

  // Cooperative-load assignment: 128 rows x 8 float4 per chunk = 1024 float4,
  // 256 threads -> 4 each. Gather base pointers hoisted out of the K loop.
  int ar[4], ac[4];
  const float* asrc[4];
  const float* bsrc[4];
#pragma unroll
  for (int i = 0; i < 4; ++i) {
    int lin = tid + i * 256;
    int r = lin >> 3, c4 = (lin & 7) << 2;
    ar[i] = r; ac[i] = c4;
    asrc[i] = input    + (size_t)wbi[c * TPC + r] * NHID + c4;
    bsrc[i] = word_emb + (size_t)cluster[c * CSIZE + n0 + r] * NHID + c4;
  }

  v8f acc[2][4];
#pragma unroll
  for (int mt = 0; mt < 2; ++mt)
#pragma unroll
    for (int nt = 0; nt < 4; ++nt)
      acc[mt][nt] = (v8f){0.f, 0.f, 0.f, 0.f, 0.f, 0.f, 0.f, 0.f};

  for (int k0 = 0; k0 < NHID; k0 += KC) {
#pragma unroll
    for (int i = 0; i < 4; ++i) {
      float4 av = *(const float4*)(asrc[i] + k0);
      float4 bv = *(const float4*)(bsrc[i] + k0);
      int off = ar[i] * LDA + ac[i];
      cvt_store(Ahi, Alo, off, av);
      cvt_store(Bhi, Blo, off, bv);
    }
    __syncthreads();

    v16bf ah[2], al[2], bh[4], bl[4];
#pragma unroll
    for (int mt = 0; mt < 2; ++mt) {
      int row = wm * 32 + mt * 16 + (lane & 15);
      ah[mt] = frag_ld(Ahi, row, lane);
      al[mt] = frag_ld(Alo, row, lane);
    }
#pragma unroll
    for (int nt = 0; nt < 4; ++nt) {
      int row = wn * 64 + nt * 16 + (lane & 15);
      bh[nt] = frag_ld(Bhi, row, lane);
      bl[nt] = frag_ld(Blo, row, lane);
    }
#pragma unroll
    for (int mt = 0; mt < 2; ++mt)
#pragma unroll
      for (int nt = 0; nt < 4; ++nt) {
        acc[mt][nt] = wmma_bf16(ah[mt], bh[nt], acc[mt][nt]);  // hi*hi
        acc[mt][nt] = wmma_bf16(ah[mt], bl[nt], acc[mt][nt]);  // hi*lo
        acc[mt][nt] = wmma_bf16(al[mt], bh[nt], acc[mt][nt]);  // lo*hi
      }
    __syncthreads();
  }

  // Epilogue: C/D layout -> VGPR r holds (M = r + 8*(lane>>4), N = lane&15)
#pragma unroll
  for (int nt = 0; nt < 4; ++nt) {
    int n = wn * 64 + nt * 16 + (lane & 15);
    float bias = word_bias[cluster[c * CSIZE + n0 + n]];
#pragma unroll
    for (int mt = 0; mt < 2; ++mt) {
#pragma unroll
      for (int r = 0; r < 8; ++r) {
        int m = wm * 32 + mt * 16 + r + ((lane >> 4) << 3);
        out_words[((size_t)c * TPC + m) * CSIZE + n0 + n] = acc[mt][nt][r] + bias;
      }
    }
  }
}

// ---------------------------------------------------------------------------
// p_class = input @ cls_w.T + cls_b.  M=8192, N=64, K=1024.
// grid = NTOK/128 blocks, 256 threads (8 waves); wave w -> 16-row strip.
// ---------------------------------------------------------------------------
__global__ __launch_bounds__(256) void class_gemm_kernel(
    const float* __restrict__ input, const float* __restrict__ cls_w,
    const float* __restrict__ cls_b, float* __restrict__ out_class)
{
  __shared__ __bf16 Ahi[128 * LDA], Alo[128 * LDA];
  __shared__ __bf16 Bhi[NCLS * LDA], Blo[NCLS * LDA];

  const int m0   = blockIdx.x * 128;
  const int tid  = threadIdx.x;
  const int lane = tid & 31;
  const int wave = tid >> 5;

  v8f acc[4];
#pragma unroll
  for (int nt = 0; nt < 4; ++nt)
    acc[nt] = (v8f){0.f, 0.f, 0.f, 0.f, 0.f, 0.f, 0.f, 0.f};

  for (int k0 = 0; k0 < NHID; k0 += KC) {
    // A tile: 128 rows x 8 float4 = 1024 -> 4 per thread
#pragma unroll
    for (int i = 0; i < 4; ++i) {
      int lin = tid + i * 256;
      int r = lin >> 3, c4 = (lin & 7) << 2;
      float4 av = *(const float4*)(input + (size_t)(m0 + r) * NHID + k0 + c4);
      cvt_store(Ahi, Alo, r * LDA + c4, av);
    }
    // B tile: 64 rows x 8 float4 = 512 -> 2 per thread
#pragma unroll
    for (int i = 0; i < 2; ++i) {
      int lin = tid + i * 256;
      int r = lin >> 3, c4 = (lin & 7) << 2;
      float4 bv = *(const float4*)(cls_w + (size_t)r * NHID + k0 + c4);
      cvt_store(Bhi, Blo, r * LDA + c4, bv);
    }
    __syncthreads();

    int arow = wave * 16 + (lane & 15);
    v16bf ah = frag_ld(Ahi, arow, lane);
    v16bf al = frag_ld(Alo, arow, lane);
#pragma unroll
    for (int nt = 0; nt < 4; ++nt) {
      int brow = nt * 16 + (lane & 15);
      v16bf bh = frag_ld(Bhi, brow, lane);
      v16bf bl = frag_ld(Blo, brow, lane);
      acc[nt] = wmma_bf16(ah, bh, acc[nt]);
      acc[nt] = wmma_bf16(ah, bl, acc[nt]);
      acc[nt] = wmma_bf16(al, bh, acc[nt]);
    }
    __syncthreads();
  }

#pragma unroll
  for (int nt = 0; nt < 4; ++nt) {
    int n = nt * 16 + (lane & 15);
    float bias = cls_b[n];
#pragma unroll
    for (int r = 0; r < 8; ++r) {
      int m = wave * 16 + r + ((lane >> 4) << 3);
      out_class[(size_t)(m0 + m) * NCLS + n] = acc[nt][r] + bias;
    }
  }
}

extern "C" void kernel_launch(void* const* d_in, const int* in_sizes, int n_in,
                              void* d_out, int out_size, void* d_ws, size_t ws_size,
                              hipStream_t stream) {
  const float* input     = (const float*)d_in[0];
  const float* cls_w     = (const float*)d_in[1];
  const float* cls_b     = (const float*)d_in[2];
  const float* word_emb  = (const float*)d_in[3];
  const float* word_bias = (const float*)d_in[4];
  const int*   wbi       = (const int*)d_in[5];
  const int*   cluster   = (const int*)d_in[6];

  float* out_class = (float*)d_out;                         // [NTOK, NCLS]
  float* out_words = out_class + (size_t)NTOK * NCLS;       // [NCLS, TPC, CSIZE]

  class_gemm_kernel<<<dim3(NTOK / 128), dim3(256), 0, stream>>>(
      input, cls_w, cls_b, out_class);

  words_gemm_kernel<<<dim3(CSIZE / 128, NCLS), dim3(256), 0, stream>>>(
      input, word_emb, word_bias, wbi, cluster, out_words);
}
